// PaiNN_49684181680725
// MI455X (gfx1250) — compile-verified, matrix operand order checked by reading, not run on hardware
//
#include <hip/hip_runtime.h>

#define NATOMS 25000
#define NEDGES 200000
#define FDIM   128
#define NINT   3
#define NRBF   20
#define NFILT  (3 * FDIM * NINT)   // 1152
#define CUT    5.0f

typedef __attribute__((ext_vector_type(16))) _Float16 v16h;
typedef __attribute__((ext_vector_type(8)))  float    v8f;

// silu via hardware v_rcp_f32 (avoids the IEEE div_scale/fma fixup chain;
// inputs came through f16 WMMA so 1-ulp rcp is already beyond data precision)
__device__ __forceinline__ float silu_f(float v) {
  return v * __builtin_amdgcn_rcpf(1.0f + __expf(-v));
}

// Load a 16x32 f16 A-fragment (or 32x16 B-fragment from a [N][K]-transposed
// weight matrix) per the CDNA5 16-bit A layout: lane&15 = row, lane>>4 picks
// the K-half; each lane holds two contiguous 8-element runs (k, k+16).
__device__ __forceinline__ v16h load_frag(const _Float16* base, int ld, int row0,
                                          int kbase, int lane) {
  int m = lane & 15, kh = lane >> 4;
  const _Float16* p = base + (size_t)(row0 + m) * ld + kbase + kh * 8;
  v16h r;
#pragma unroll
  for (int e = 0; e < 8; ++e) { r[e] = p[e]; r[e + 8] = p[e + 16]; }
  return r;
}

// ---------------- one-time prep kernels ----------------

// fp32 [T][K][N] -> f16 [T][N][K]
__global__ void k_wconv(const float* __restrict__ src, _Float16* __restrict__ dst,
                        int T, int K, int Nc) {
  long total = (long)T * K * Nc;
  for (long i = blockIdx.x * (long)blockDim.x + threadIdx.x; i < total;
       i += (long)gridDim.x * blockDim.x) {
    int t   = (int)(i / ((long)K * Nc));
    int rem = (int)(i - (long)t * K * Nc);
    int k = rem / Nc, n = rem % Nc;
    dst[(long)t * K * Nc + (long)n * K + k] = (_Float16)src[i];
  }
}

// fW [20][1152] -> fWt [1152][20] (fp32, for per-lane contiguous GEMV reads)
__global__ void k_fwt(const float* __restrict__ fW, float* __restrict__ fWt) {
  int total = NRBF * NFILT;
  for (int i = blockIdx.x * blockDim.x + threadIdx.x; i < total;
       i += gridDim.x * blockDim.x) {
    int r = i / NFILT, c = i % NFILT;
    fWt[c * NRBF + r] = fW[i];
  }
}

// q = emb[z]; mu = 0; dmu = 0
__global__ void k_init(const int* __restrict__ z, const float* __restrict__ emb,
                       float* __restrict__ q, float* __restrict__ mu,
                       float* __restrict__ dmu) {
  long total = (long)NATOMS * 384;
  for (long i = blockIdx.x * (long)blockDim.x + threadIdx.x; i < total;
       i += (long)gridDim.x * blockDim.x) {
    mu[i] = 0.0f; dmu[i] = 0.0f;
    if (i < (long)NATOMS * FDIM) {
      int a = (int)(i / FDIM), f = (int)(i % FDIM);
      q[i] = emb[(long)z[a] * FDIM + f];
    }
  }
}

// per-edge: dir(3), fcut(1), phi(20)  -> egeo[E][24]
__global__ void k_edgegeo(const float* __restrict__ pos, const int* __restrict__ idx_i,
                          const int* __restrict__ idx_j, float* __restrict__ egeo) {
  int e = blockIdx.x * blockDim.x + threadIdx.x;
  if (e >= NEDGES) return;
  int i = idx_i[e], j = idx_j[e];
  float rx = pos[j * 3 + 0] - pos[i * 3 + 0];
  float ry = pos[j * 3 + 1] - pos[i * 3 + 1];
  float rz = pos[j * 3 + 2] - pos[i * 3 + 2];
  rx = (fabsf(rx) < 1e-6f) ? 1e-6f : rx;
  ry = (fabsf(ry) < 1e-6f) ? 1e-6f : ry;
  rz = (fabsf(rz) < 1e-6f) ? 1e-6f : rz;
  float d2 = rx * rx + ry * ry + rz * rz;
  float d = sqrtf(d2);
  float inv = __builtin_amdgcn_rcpf(d);   // v_rcp_f32
  float* g = egeo + (size_t)e * 24;
  g[0] = rx * inv; g[1] = ry * inv; g[2] = rz * inv;
  g[3] = (d < CUT) ? 0.5f * (__cosf(3.14159265358979f * d / CUT) + 1.0f) : 0.0f;
  const float width = CUT / (NRBF - 1);
  const float coeff = -0.5f / (width * width);
#pragma unroll
  for (int r = 0; r < NRBF; ++r) {
    float dd = d - width * (float)r;
    g[4 + r] = __expf(coeff * dd * dd);
  }
}

// ---------------- per-interaction kernels ----------------

// x = silu(q @ iW1[t] + ib1[t]) @ iW2[t] + ib2[t]   (block = 16 atoms, 8 waves)
__global__ __launch_bounds__(256) void k_atom_mlp1(
    const float* __restrict__ q, const _Float16* __restrict__ w1t,
    const float* __restrict__ ib1, const _Float16* __restrict__ w2t,
    const float* __restrict__ ib2, float* __restrict__ x, int t) {
  __shared__ _Float16 sQ[16 * FDIM];
  __shared__ _Float16 sH[16 * FDIM];
  int tid = threadIdx.x, lane = tid & 31, wave = tid >> 5;
  int atom0 = blockIdx.x * 16;
  const _Float16* W1 = w1t + (size_t)t * FDIM * FDIM;       // [128][128]
  const _Float16* W2 = w2t + (size_t)t * 3 * FDIM * FDIM;   // [384][128]

  for (int i = tid; i < 16 * FDIM; i += 256) {
    int row = i >> 7, col = i & 127, a = atom0 + row;
    sQ[i] = (_Float16)((a < NATOMS) ? q[(size_t)a * FDIM + col] : 0.0f);
  }
  __syncthreads();

  { // layer 1, ntile = wave
    int col0 = wave * 16;
    float b = ib1[t * FDIM + col0 + (lane & 15)];
    v8f acc;
#pragma unroll
    for (int r = 0; r < 8; ++r) acc[r] = b;
#pragma unroll
    for (int kc = 0; kc < 4; ++kc) {
      v16h a  = load_frag(sQ, FDIM, 0, kc * 32, lane);
      v16h bb = load_frag(W1, FDIM, col0, kc * 32, lane);
      acc = __builtin_amdgcn_wmma_f32_16x16x32_f16(false, a, false, bb, (short)0,
                                                   acc, false, false);
    }
    int n = lane & 15, kh = lane >> 4;
#pragma unroll
    for (int r = 0; r < 8; ++r)
      sH[(kh * 8 + r) * FDIM + col0 + n] = (_Float16)silu_f(acc[r]);
  }
  __syncthreads();

  for (int ti = 0; ti < 3; ++ti) { // layer 2, 24 ntiles
    int nt = wave + ti * 8, col0 = nt * 16;
    float b = ib2[t * 3 * FDIM + col0 + (lane & 15)];
    v8f acc;
#pragma unroll
    for (int r = 0; r < 8; ++r) acc[r] = b;
#pragma unroll
    for (int kc = 0; kc < 4; ++kc) {
      v16h a  = load_frag(sH, FDIM, 0, kc * 32, lane);
      v16h bb = load_frag(W2, FDIM, col0, kc * 32, lane);
      acc = __builtin_amdgcn_wmma_f32_16x16x32_f16(false, a, false, bb, (short)0,
                                                   acc, false, false);
    }
    int n = lane & 15, kh = lane >> 4;
#pragma unroll
    for (int r = 0; r < 8; ++r) {
      int a2 = atom0 + kh * 8 + r;
      if (a2 < NATOMS) x[(size_t)a2 * 384 + col0 + n] = acc[r];
    }
  }
}

// one wave per edge; lane owns 4 features; atomic scatter into q and dmu
__global__ __launch_bounds__(256) void k_edge_msg(
    const int* __restrict__ idx_i, const int* __restrict__ idx_j,
    const float* __restrict__ egeo, const float* __restrict__ fWt,
    const float* __restrict__ fb, const float* __restrict__ x,
    const float* __restrict__ mu, float* __restrict__ q,
    float* __restrict__ dmu, int t) {
  int wave = threadIdx.x >> 5, lane = threadIdx.x & 31;
  int e = blockIdx.x * 8 + wave;
  if (e >= NEDGES) return;
  int ai = idx_i[e], aj = idx_j[e];
  const float* g = egeo + (size_t)e * 24;
  float dir0 = g[0], dir1 = g[1], dir2 = g[2], fc = g[3];
  float ph[NRBF];
#pragma unroll
  for (int r = 0; r < NRBF; ++r) ph[r] = g[4 + r];
  const float* xj = x + (size_t)aj * 384;
  const float* mj = mu + (size_t)aj * 384;
  float* di = dmu + (size_t)ai * 384;
#pragma unroll
  for (int u = 0; u < 4; ++u) {
    int f = lane * 4 + u;
    float fl[3];
#pragma unroll
    for (int p = 0; p < 3; ++p) {
      int col = t * 384 + p * 128 + f;
      const float* w = fWt + col * NRBF;
      float s = fb[col];
#pragma unroll
      for (int r = 0; r < NRBF; ++r) s += ph[r] * w[r];
      fl[p] = s * fc;
    }
    atomicAdd(q + (size_t)ai * 128 + f, fl[0] * xj[f]);
    float dR = fl[1] * xj[128 + f];
    float dM = fl[2] * xj[256 + f];
    atomicAdd(di + f,       dR * dir0 + dM * mj[f]);
    atomicAdd(di + 128 + f, dR * dir1 + dM * mj[128 + f]);
    atomicAdd(di + 256 + f, dR * dir2 + dM * mj[256 + f]);
  }
}

// apply dmu, then intra-atomic mixing (block = 16 atoms, 8 waves, 60KB LDS)
__global__ __launch_bounds__(256) void k_mix(
    float* __restrict__ q, float* __restrict__ mu, float* __restrict__ dmu,
    const _Float16* __restrict__ wmixT, const _Float16* __restrict__ w1T,
    const float* __restrict__ mb1, const _Float16* __restrict__ w2T,
    const float* __restrict__ mb2, int t) {
  __shared__ _Float16 sA[48 * FDIM];    // muA[48][128] -> ctx[16][256] -> h[16][128]
  __shared__ _Float16 sMix[48 * 256];   // mu_mix (f16)
  __shared__ float    sX2[16 * 384];    // MLP output (f32)
  int tid = threadIdx.x, lane = tid & 31, wave = tid >> 5;
  int atom0 = blockIdx.x * 16;
  const _Float16* Wmix = wmixT + (size_t)t * 256 * FDIM;    // [256][128]
  const _Float16* W1   = w1T + (size_t)t * FDIM * 256;      // [128][256]
  const _Float16* W2   = w2T + (size_t)t * 3 * FDIM * FDIM; // [384][128]

  // phase 0: mu_new = mu + dmu; commit mu; zero dmu; stage A tile
  for (int i = tid; i < 48 * FDIM; i += 256) {
    int row = i >> 7, col = i & 127;
    int a = atom0 + row / 3, d = row - (row / 3) * 3;
    float v = 0.0f;
    if (a < NATOMS) {
      size_t gi = (size_t)a * 384 + d * 128 + col;
      v = mu[gi] + dmu[gi];
      mu[gi] = v;
      dmu[gi] = 0.0f;
    }
    sA[i] = (_Float16)v;
  }
  __syncthreads();

  // phase 1: mu_mix[48][256] = muA @ Wmix (no bias)
  for (int ti = 0; ti < 6; ++ti) {
    int tt = wave + ti * 8;            // 0..47
    int mt = tt >> 4, nt = tt & 15;
    v8f acc;
#pragma unroll
    for (int r = 0; r < 8; ++r) acc[r] = 0.0f;
#pragma unroll
    for (int kc = 0; kc < 4; ++kc) {
      v16h a  = load_frag(sA, FDIM, mt * 16, kc * 32, lane);
      v16h bb = load_frag(Wmix, FDIM, nt * 16, kc * 32, lane);
      acc = __builtin_amdgcn_wmma_f32_16x16x32_f16(false, a, false, bb, (short)0,
                                                   acc, false, false);
    }
    int n = lane & 15, kh = lane >> 4;
#pragma unroll
    for (int r = 0; r < 8; ++r)
      sMix[(mt * 16 + kh * 8 + r) * 256 + nt * 16 + n] = (_Float16)acc[r];
  }
  __syncthreads();

  // phase 2: ctx[16][256] = [q | ||mu_V||]  into sA
  for (int i = tid; i < 16 * 256; i += 256) {
    int a = i >> 8, c = i & 255;
    float v;
    if (c < 128) {
      int atom = atom0 + a;
      v = (atom < NATOMS) ? q[(size_t)atom * 128 + c] : 0.0f;
    } else {
      int f = c - 128;
      float s = 1e-8f;
#pragma unroll
      for (int d = 0; d < 3; ++d) {
        float mv = (float)sMix[(a * 3 + d) * 256 + f];
        s += mv * mv;
      }
      v = sqrtf(s);
    }
    sA[i] = (_Float16)v;
  }
  __syncthreads();

  // phase 3: h = silu(ctx @ mW1 + mb1); K=256
  {
    int col0 = wave * 16;
    float b = mb1[t * 128 + col0 + (lane & 15)];
    v8f acc;
#pragma unroll
    for (int r = 0; r < 8; ++r) acc[r] = b;
#pragma unroll
    for (int kc = 0; kc < 8; ++kc) {
      v16h a  = load_frag(sA, 256, 0, kc * 32, lane);
      v16h bb = load_frag(W1, 256, col0, kc * 32, lane);
      acc = __builtin_amdgcn_wmma_f32_16x16x32_f16(false, a, false, bb, (short)0,
                                                   acc, false, false);
    }
    __syncthreads();   // everyone done reading ctx before h overwrites sA
    int n = lane & 15, kh = lane >> 4;
#pragma unroll
    for (int r = 0; r < 8; ++r)
      sA[(kh * 8 + r) * 128 + col0 + n] = (_Float16)silu_f(acc[r]);
  }
  __syncthreads();

  // phase 4: x2[16][384] = h @ mW2 + mb2
  for (int ti = 0; ti < 3; ++ti) {
    int nt = wave + ti * 8, col0 = nt * 16;
    float b = mb2[t * 384 + col0 + (lane & 15)];
    v8f acc;
#pragma unroll
    for (int r = 0; r < 8; ++r) acc[r] = b;
#pragma unroll
    for (int kc = 0; kc < 4; ++kc) {
      v16h a  = load_frag(sA, FDIM, 0, kc * 32, lane);
      v16h bb = load_frag(W2, FDIM, col0, kc * 32, lane);
      acc = __builtin_amdgcn_wmma_f32_16x16x32_f16(false, a, false, bb, (short)0,
                                                   acc, false, false);
    }
    int n = lane & 15, kh = lane >> 4;
#pragma unroll
    for (int r = 0; r < 8; ++r)
      sX2[(kh * 8 + r) * 384 + col0 + n] = acc[r];
  }
  __syncthreads();

  // phase 5: gated updates of q and mu
  for (int i = tid; i < 16 * 128; i += 256) {
    int a = i >> 7, f = i & 127;
    int atom = atom0 + a;
    if (atom >= NATOMS) continue;
    float s = 0.0f, mw[3];
#pragma unroll
    for (int d = 0; d < 3; ++d) {
      float mv = (float)sMix[(a * 3 + d) * 256 + f];
      mw[d] = (float)sMix[(a * 3 + d) * 256 + 128 + f];
      s += mv * mw[d];
    }
    q[(size_t)atom * 128 + f] += sX2[a * 384 + f] + sX2[a * 384 + 256 + f] * s;
    float dmf = sX2[a * 384 + 128 + f];
#pragma unroll
    for (int d = 0; d < 3; ++d)
      mu[(size_t)atom * 384 + d * 128 + f] += dmf * mw[d];
  }
}

// ---------------- host launcher ----------------

extern "C" void kernel_launch(void* const* d_in, const int* in_sizes, int n_in,
                              void* d_out, int out_size, void* d_ws, size_t ws_size,
                              hipStream_t stream) {
  (void)in_sizes; (void)n_in; (void)out_size; (void)ws_size;
  const int*   z     = (const int*)  d_in[0];
  const float* pos   = (const float*)d_in[1];
  const int*   idx_i = (const int*)  d_in[2];
  const int*   idx_j = (const int*)  d_in[3];
  const float* emb   = (const float*)d_in[4];
  const float* fW    = (const float*)d_in[5];
  const float* fb    = (const float*)d_in[6];
  const float* iW1   = (const float*)d_in[7];
  const float* ib1   = (const float*)d_in[8];
  const float* iW2   = (const float*)d_in[9];
  const float* ib2   = (const float*)d_in[10];
  const float* mWmix = (const float*)d_in[11];
  const float* mW1   = (const float*)d_in[12];
  const float* mb1   = (const float*)d_in[13];
  const float* mW2   = (const float*)d_in[14];
  const float* mb2   = (const float*)d_in[15];

  float* q  = (float*)d_out;                  // [N,128]
  float* mu = q + (size_t)NATOMS * 128;       // [N,3,128]

  char* ws = (char*)d_ws;
  size_t off = 0;
  auto take = [&](size_t bytes) -> char* {
    char* p = ws + off;
    off = (off + bytes + 255) & ~(size_t)255;
    return p;
  };
  float* xbuf = (float*)take((size_t)NATOMS * 384 * 4);
  float* dmu  = (float*)take((size_t)NATOMS * 384 * 4);
  float* egeo = (float*)take((size_t)NEDGES * 24 * 4);
  float* fWt  = (float*)take((size_t)NFILT * NRBF * 4);
  _Float16* w16 = (_Float16*)take((size_t)540672 * 2);
  _Float16* iW1t   = w16;            // 3 x [128][128]
  _Float16* iW2t   = w16 + 49152;    // 3 x [384][128]
  _Float16* mWmixT = w16 + 196608;   // 3 x [256][128]
  _Float16* mW1T   = w16 + 294912;   // 3 x [128][256]
  _Float16* mW2T   = w16 + 393216;   // 3 x [384][128]

  k_wconv<<<256, 256, 0, stream>>>(iW1,   iW1t,   3, 128, 128);
  k_wconv<<<256, 256, 0, stream>>>(iW2,   iW2t,   3, 128, 384);
  k_wconv<<<256, 256, 0, stream>>>(mWmix, mWmixT, 3, 128, 256);
  k_wconv<<<256, 256, 0, stream>>>(mW1,   mW1T,   3, 256, 128);
  k_wconv<<<256, 256, 0, stream>>>(mW2,   mW2T,   3, 128, 384);
  k_fwt<<<96, 256, 0, stream>>>(fW, fWt);
  k_init<<<4096, 256, 0, stream>>>(z, emb, q, mu, dmu);
  k_edgegeo<<<(NEDGES + 255) / 256, 256, 0, stream>>>(pos, idx_i, idx_j, egeo);

  int ablocks = (NATOMS + 15) / 16;
  int eblocks = (NEDGES + 7) / 8;
  for (int t = 0; t < NINT; ++t) {
    k_atom_mlp1<<<ablocks, 256, 0, stream>>>(q, iW1t, ib1, iW2t, ib2, xbuf, t);
    k_edge_msg<<<eblocks, 256, 0, stream>>>(idx_i, idx_j, egeo, fWt, fb, xbuf,
                                            mu, q, dmu, t);
    k_mix<<<ablocks, 256, 0, stream>>>(q, mu, dmu, mWmixT, mW1T, mb1, mW2T,
                                       mb2, t);
  }
}
